// CausalTransformerDynamics_69148973466387
// MI455X (gfx1250) — matrix operand order repaired and checked by reading, not compiled, not run on hardware
//
#include <hip/hip_runtime.h>
#include <hip/hip_bf16.h>
#include <hip/hip_fp16.h>

// ---- problem constants (match reference) ----
#define TDIM   1024
#define DMODEL 1024
#define NHEAD  16
#define DHEAD  64
#define NLAYER 4
#define PDIM   256
#define SCDIM  512
#define SDDIM  256
#define BATCH  4
#define DFFDIM 4096
#define ROWS   (BATCH * TDIM)   // 4096
#define QKVN   (3 * DMODEL)     // 3072

typedef __attribute__((ext_vector_type(16))) _Float16 v16h;
typedef __attribute__((ext_vector_type(8)))  float    v8f;

// ---------------- WMMA fragment loaders (all f16 now) ----------------
// A-matrix 16x32 f16 layout: lane = m (0..15) per half;
// lanes 0-15 hold K = 0..15 contiguous, lanes 16-31 hold K = 16..31 contiguous.
// caller passes p already offset by row*stride + kk + 16*(lane>>4).
__device__ __forceinline__ v16h fragA_f16(const _Float16* __restrict__ p) {
  v16h f;
#pragma unroll
  for (int i = 0; i < 8; ++i) f[i] = p[i];
#pragma unroll
  for (int i = 0; i < 8; ++i) f[8 + i] = p[8 + i];
  return f;
}
// B-matrix 32x16 f16 layout: lane = n (0..15) per half; lanes 0-15 hold
// K=0..15 contiguous, lanes 16-31 hold K=16..31 contiguous.
// caller passes p already offset by n*stride + kk + 16*(lane>>4).
__device__ __forceinline__ v16h fragB_f16(const _Float16* __restrict__ p) {
  v16h f;
#pragma unroll
  for (int i = 0; i < 16; ++i) f[i] = p[i];
  return f;
}

// ---------------- f32 -> f16 conversion (weights, final x) ----------------
__global__ __launch_bounds__(256) void cvt_f16_kernel(
    const float* __restrict__ src, _Float16* __restrict__ dst, int n)
{
  const int i = blockIdx.x * 256 + threadIdx.x;
  if (i < n) dst[i] = (_Float16)src[i];
}

// ---------------- cross-speaker causal context (O(T*PD) scan) ----------------
__global__ __launch_bounds__(256) void cross_ctx_kernel(
    const float* __restrict__ delta, const unsigned char* __restrict__ valid,
    const int* __restrict__ spk, float* __restrict__ cross)
{
  const int b = blockIdx.x;
  const int p = threadIdx.x;                 // PD = 256
  __shared__ float accTot[PDIM];
  __shared__ float accSpk[16][PDIM];
  __shared__ float cnts[17];                 // [0..15] per-speaker, [16] total
  accTot[p] = 0.f;
#pragma unroll
  for (int s = 0; s < 16; ++s) accSpk[s][p] = 0.f;
  if (p < 17) cnts[p] = 0.f;
  __syncthreads();
  for (int t = 0; t < TDIM; ++t) {
    int ss = spk[b * TDIM + t];
    ss = ss < 0 ? 0 : (ss > 15 ? 15 : ss);
    const float vf = valid[b * TDIM + t] ? 1.f : 0.f;
    const float denom = fmaxf(cnts[16] - cnts[ss], 1e-9f);
    cross[((size_t)b * TDIM + t) * PDIM + p] = (accTot[p] - accSpk[ss][p]) / denom;
    const float w = vf * delta[((size_t)b * TDIM + t) * PDIM + p];
    __syncthreads();                          // everyone has read cnts/acc
    accTot[p] += w;
    accSpk[ss][p] += w;
    if (p == 0) { cnts[ss] += vf; cnts[16] += vf; }
    __syncthreads();                          // updates visible next step
  }
}

// ---------------- pack [delta | sctx | cross] -> Xin16 [ROWS, 1024] f16 -------
__global__ __launch_bounds__(256) void pack_kernel(
    const float* __restrict__ delta, const float* __restrict__ sctx,
    const float* __restrict__ cross, _Float16* __restrict__ xin)
{
  const int row = blockIdx.x;
  for (int c = threadIdx.x; c < DMODEL; c += 256) {
    float v;
    if (c < PDIM)               v = delta[(size_t)row * PDIM + c];
    else if (c < PDIM + SCDIM)  v = sctx [(size_t)row * SCDIM + (c - PDIM)];
    else                        v = cross[(size_t)row * PDIM  + (c - PDIM - SCDIM)];
    xin[(size_t)row * DMODEL + c] = (_Float16)v;
  }
}

// ---------------- WMMA f16 GEMM: C = act(A @ W^T + bias) (+residual) ---------
// A: [M,K] f16 row-major, W: [N,K] f16 row-major.
// Block = 8 waves (2M x 4N); wave tile = 32(M) x 64(N); block tile 64 x 256.
// Requires M%64==0, N%256==0, K%32==0. Cf (f32) and/or Ch (f16) outputs.
__global__ __launch_bounds__(256) void gemm_kernel(
    const _Float16* __restrict__ A, const _Float16* __restrict__ W,
    const float* __restrict__ bias, const float* __restrict__ residual,
    float* __restrict__ Cf, _Float16* __restrict__ Ch,
    int M, int N, int K, int relu)
{
  const int lane = threadIdx.x & 31;
  const int wave = threadIdx.x >> 5;
  const int l15  = lane & 15;
  const int hi   = lane >> 4;
  const int aoff = hi << 4;    // A chunk base: 16*hi
  const int boff = hi << 4;    // B chunk base: 16*hi
  const int m0 = blockIdx.x * 64 + (wave & 1) * 32;
  const int n0 = blockIdx.y * 256 + (wave >> 1) * 64;

  v8f acc[2][4];
#pragma unroll
  for (int mi = 0; mi < 2; ++mi)
#pragma unroll
    for (int t = 0; t < 4; ++t)
#pragma unroll
      for (int r = 0; r < 8; ++r) acc[mi][t][r] = 0.f;

  const _Float16* arow0 = A + (size_t)(m0 + l15) * K + aoff;
  const _Float16* arow1 = A + (size_t)(m0 + 16 + l15) * K + aoff;
  for (int kk = 0; kk < K; kk += 32) {
    const v16h a0 = fragA_f16(arow0 + kk);
    const v16h a1 = fragA_f16(arow1 + kk);
#pragma unroll
    for (int t = 0; t < 4; ++t) {
      const v16h bf = fragB_f16(W + (size_t)(n0 + t * 16 + l15) * K + kk + boff);
      acc[0][t] = __builtin_amdgcn_wmma_f32_16x16x32_f16(
          false, a0, false, bf, (short)0, acc[0][t], false, false);
      acc[1][t] = __builtin_amdgcn_wmma_f32_16x16x32_f16(
          false, a1, false, bf, (short)0, acc[1][t], false, false);
    }
  }
#pragma unroll
  for (int mi = 0; mi < 2; ++mi) {
#pragma unroll
    for (int t = 0; t < 4; ++t) {
      const int n = n0 + t * 16 + l15;
      const float bv = bias ? bias[n] : 0.f;
#pragma unroll
      for (int r = 0; r < 8; ++r) {
        const int m = m0 + mi * 16 + r + 8 * hi;
        float v = acc[mi][t][r] + bv;
        if (relu) v = fmaxf(v, 0.f);
        if (residual) v += residual[(size_t)m * N + n];
        if (Cf) Cf[(size_t)m * N + n] = v;
        if (Ch) Ch[(size_t)m * N + n] = (_Float16)v;
      }
    }
  }
}

// ---------------- LayerNorm over last dim, f16 output ----------------
__global__ __launch_bounds__(256) void ln_kernel(
    const float* __restrict__ x, const float* __restrict__ g,
    const float* __restrict__ bb, _Float16* __restrict__ out, int d)
{
  const int row = blockIdx.x;
  const int tid = threadIdx.x;
  const float* xr = x + (size_t)row * d;
  float s = 0.f, s2 = 0.f;
  for (int c = tid; c < d; c += 256) { const float v = xr[c]; s += v; s2 += v * v; }
  __shared__ float r0[256], r1[256];
  r0[tid] = s; r1[tid] = s2; __syncthreads();
  for (int off = 128; off > 0; off >>= 1) {
    if (tid < off) { r0[tid] += r0[tid + off]; r1[tid] += r1[tid + off]; }
    __syncthreads();
  }
  const float mean = r0[0] / (float)d;
  const float var  = r1[0] / (float)d - mean * mean;
  const float inv  = rsqrtf(var + 1e-5f);
  _Float16* orow = out + (size_t)row * d;
  for (int c = tid; c < d; c += 256)
    orow[c] = (_Float16)((xr[c] - mean) * inv * g[c] + bb[c]);
}

// ---------------- V transpose: qkv16 [ROWS,3072] -> Vt [B,H,64,T] f16 ---------
__global__ __launch_bounds__(256) void vtrans_kernel(
    const _Float16* __restrict__ qkv, _Float16* __restrict__ Vt)
{
  const int idx = blockIdx.x * 256 + threadIdx.x;   // ROWS*DMODEL
  const int c   = idx & (DMODEL - 1);
  const int row = idx >> 10;
  const int t   = row & (TDIM - 1);
  const int b   = row >> 10;
  const int h   = c >> 6;
  const int d   = c & 63;
  Vt[(((size_t)b * NHEAD + h) * DHEAD + d) * TDIM + t] =
      qkv[(size_t)row * QKVN + 2 * DMODEL + c];
}

// ---------------- flash attention (ALiBi + causal + key padding) --------------
// One wave per 16-query block; online softmax over 32-key chunks.
// Q/K fragments are read directly from the interleaved qkv16 buffer.
__global__ __launch_bounds__(256) void attn_kernel(
    const _Float16* __restrict__ qkv, const _Float16* __restrict__ Vt,
    const unsigned char* __restrict__ valid, _Float16* __restrict__ attn_out)
{
  __shared__ _Float16 Pl[8][16][32];          // per-wave P staging (C->A relayout)
  const int lane = threadIdx.x & 31;
  const int wave = threadIdx.x >> 5;
  const int l15  = lane & 15;
  const int hi   = lane >> 4;
  const int aoff = hi << 4;
  const int boff = hi << 4;

  const int gid = blockIdx.x * 8 + wave;      // over B*H*(T/16) = 4096
  const int ib = gid & 63;
  const int h  = (gid >> 6) & 15;
  const int b  = gid >> 10;
  const int i0 = ib * 16;

  const size_t bh = (size_t)b * NHEAD + h;
  // Q rows: qkv[(b*T + i)*3072 + h*64 + d]
  const _Float16* qbase = qkv + (size_t)(b * TDIM + i0 + l15) * QKVN + h * DHEAD + aoff;
  const v16h qf0 = fragA_f16(qbase);
  const v16h qf1 = fragA_f16(qbase + 32);

  const float slope = exp2f(-0.5f * (float)(h + 1));   // H=16 power-of-two path
  const unsigned char* vmask = valid + b * TDIM;

  float m8[8], s8[8];
  v8f o[4];
#pragma unroll
  for (int r = 0; r < 8; ++r) { m8[r] = -1e30f; s8[r] = 0.f; }
#pragma unroll
  for (int dt = 0; dt < 4; ++dt)
#pragma unroll
    for (int r = 0; r < 8; ++r) o[dt][r] = 0.f;

  for (int jb = 0; jb < i0 + 16; jb += 32) {
    // ---- S = Q @ K^T for 32 keys (two 16-wide tiles) ----
    v8f s[2];
#pragma unroll
    for (int t = 0; t < 2; ++t) {
      const _Float16* kb = qkv + (size_t)(b * TDIM + jb + 16 * t + l15) * QKVN
                               + DMODEL + h * DHEAD + boff;
      const v16h kf0 = fragB_f16(kb);
      const v16h kf1 = fragB_f16(kb + 32);
      v8f z = {0.f, 0.f, 0.f, 0.f, 0.f, 0.f, 0.f, 0.f};
      z = __builtin_amdgcn_wmma_f32_16x16x32_f16(false, qf0, false, kf0, (short)0, z, false, false);
      z = __builtin_amdgcn_wmma_f32_16x16x32_f16(false, qf1, false, kf1, (short)0, z, false, false);
      s[t] = z;
    }
    // ---- mask + online softmax update ----
    float p[2][8];
#pragma unroll
    for (int r = 0; r < 8; ++r) {
      const int i = i0 + r + 8 * hi;
      float mloc = -1e30f;
#pragma unroll
      for (int t = 0; t < 2; ++t) {
        const int j = jb + 16 * t + l15;
        const float bias = (j <= i && vmask[j]) ? (-slope * (float)(i - j)) : -1e30f;
        const float sv = s[t][r] * 0.125f + bias;     // 1/sqrt(64)
        p[t][r] = sv;
        mloc = fmaxf(mloc, sv);
      }
#pragma unroll
      for (int off = 1; off < 16; off <<= 1)
        mloc = fmaxf(mloc, __shfl_xor(mloc, off, 32));
      const float mnew = fmaxf(m8[r], mloc);
      const float sc = __expf(m8[r] - mnew);
      float rs = 0.f;
#pragma unroll
      for (int t = 0; t < 2; ++t) {
        const float pv = __expf(p[t][r] - mnew);
        p[t][r] = pv;
        rs += pv;
      }
#pragma unroll
      for (int off = 1; off < 16; off <<= 1)
        rs += __shfl_xor(rs, off, 32);
      s8[r] = s8[r] * sc + rs;
      m8[r] = mnew;
#pragma unroll
      for (int dt = 0; dt < 4; ++dt) o[dt][r] *= sc;
    }
    // ---- re-lay P from C-layout to A-fragment via LDS ----
#pragma unroll
    for (int t = 0; t < 2; ++t)
#pragma unroll
      for (int r = 0; r < 8; ++r)
        Pl[wave][r + 8 * hi][16 * t + l15] = (_Float16)p[t][r];
    asm volatile("s_wait_dscnt 0" ::: "memory");
    v16h pf;
#pragma unroll
    for (int e = 0; e < 8; ++e) pf[e] = Pl[wave][l15][aoff + e];
#pragma unroll
    for (int e = 0; e < 8; ++e) pf[8 + e] = Pl[wave][l15][aoff + 8 + e];
    asm volatile("s_wait_dscnt 0" ::: "memory");
    // ---- O += P @ V (V stored transposed: [dh, T]) ----
#pragma unroll
    for (int dt = 0; dt < 4; ++dt) {
      const _Float16* vb = Vt + (bh * DHEAD + 16 * dt + l15) * TDIM + jb + boff;
      const v16h vf = fragB_f16(vb);
      o[dt] = __builtin_amdgcn_wmma_f32_16x16x32_f16(false, pf, false, vf, (short)0, o[dt], false, false);
    }
  }
  // ---- normalize + write f16 [ROWS, DMODEL] ----
#pragma unroll
  for (int r = 0; r < 8; ++r) {
    const float inv = 1.f / (s8[r] + 1e-20f);
    const int m = i0 + r + 8 * hi;
    _Float16* orow = attn_out + ((size_t)b * TDIM + m) * DMODEL + h * DHEAD;
#pragma unroll
    for (int dt = 0; dt < 4; ++dt)
      orow[16 * dt + l15] = (_Float16)(o[dt][r] * inv);
  }
}

// ---------------- final LN + shift-right-by-one-turn ----------------
__global__ __launch_bounds__(256) void out_kernel(
    const float* __restrict__ proj, const float* __restrict__ g,
    const float* __restrict__ bb, const float* __restrict__ init,
    float* __restrict__ out)
{
  const int row = blockIdx.x;               // b*T + t
  const int t = row & (TDIM - 1);
  const int j = threadIdx.x;                // SD = 256
  float* orow = out + (size_t)row * SDDIM;
  if (t == 0) { orow[j] = init[j]; return; }   // t uniform per block: safe
  const float* pr = proj + (size_t)(row - 1) * SDDIM;
  const float v = pr[j];
  __shared__ float red[256];
  red[j] = v; __syncthreads();
  for (int off = 128; off > 0; off >>= 1) { if (j < off) red[j] += red[j + off]; __syncthreads(); }
  const float mean = red[0] / 256.f;
  __syncthreads();
  const float dv = v - mean;
  red[j] = dv * dv; __syncthreads();
  for (int off = 128; off > 0; off >>= 1) { if (j < off) red[j] += red[j + off]; __syncthreads(); }
  const float inv = rsqrtf(red[0] / 256.f + 1e-5f);
  orow[j] = dv * inv * g[j] + bb[j];
}

// ---------------- host orchestration ----------------
static inline void cvt_launch(const float* src, _Float16* dst, size_t n, hipStream_t s) {
  cvt_f16_kernel<<<(unsigned)((n + 255) / 256), 256, 0, s>>>(src, dst, (int)n);
}

extern "C" void kernel_launch(void* const* d_in, const int* in_sizes, int n_in,
                              void* d_out, int out_size, void* d_ws, size_t ws_size,
                              hipStream_t stream)
{
  (void)in_sizes; (void)n_in; (void)out_size; (void)ws_size;
  const float* delta = (const float*)d_in[0];
  const float* sctx  = (const float*)d_in[1];
  const float* w_in  = (const float*)d_in[2];
  const float* b_in  = (const float*)d_in[3];
  const float* wqkv  = (const float*)d_in[4];
  const float* bqkv  = (const float*)d_in[5];
  const float* wo    = (const float*)d_in[6];
  const float* bo    = (const float*)d_in[7];
  const float* ln1g  = (const float*)d_in[8];
  const float* ln1b  = (const float*)d_in[9];
  const float* ln2g  = (const float*)d_in[10];
  const float* ln2b  = (const float*)d_in[11];
  const float* w1    = (const float*)d_in[12];
  const float* b1    = (const float*)d_in[13];
  const float* w2    = (const float*)d_in[14];
  const float* b2    = (const float*)d_in[15];
  const float* w_out = (const float*)d_in[16];
  const float* b_out = (const float*)d_in[17];
  const float* lnog  = (const float*)d_in[18];
  const float* lnob  = (const float*)d_in[19];
  const float* istate= (const float*)d_in[20];
  const unsigned char* valid = (const unsigned char*)d_in[21];
  const int* spk     = (const int*)d_in[22];
  float* outp = (float*)d_out;

  // workspace layout (~215 MB)
  char* ws = (char*)d_ws;
  size_t off = 0;
  auto take = [&](size_t bytes) -> char* {
    char* p = ws + off;
    off += (bytes + 255) & ~(size_t)255;
    return p;
  };
  float*    x0    = (float*)take((size_t)ROWS * DMODEL * 4);
  float*    x1    = (float*)take((size_t)ROWS * DMODEL * 4);
  _Float16* h16   = (_Float16*)take((size_t)ROWS * DMODEL * 2);
  _Float16* qkv16 = (_Float16*)take((size_t)ROWS * QKVN * 2);
  _Float16* Vt    = (_Float16*)take((size_t)BATCH * NHEAD * TDIM * DHEAD * 2);
  _Float16* attn16= (_Float16*)take((size_t)ROWS * DMODEL * 2);
  _Float16* ffn16 = (_Float16*)take((size_t)ROWS * DFFDIM * 2);
  float*    cross = (float*)take((size_t)ROWS * PDIM * 4);
  // f16 weights
  _Float16* w_in16 = (_Float16*)take((size_t)DMODEL * DMODEL * 2);
  _Float16* wqkv16 = (_Float16*)take((size_t)NLAYER * QKVN * DMODEL * 2);
  _Float16* wo16   = (_Float16*)take((size_t)NLAYER * DMODEL * DMODEL * 2);
  _Float16* w116   = (_Float16*)take((size_t)NLAYER * DFFDIM * DMODEL * 2);
  _Float16* w216   = (_Float16*)take((size_t)NLAYER * DMODEL * DFFDIM * 2);
  _Float16* wout16 = (_Float16*)take((size_t)SDDIM * DMODEL * 2);
  // aliases into dead regions:
  _Float16* xin16 = qkv16;               // used only before layer 0's qkv GEMM
  _Float16* x16   = attn16;              // used only after the layer loop
  float*    proj  = (float*)ffn16;       // used only after the layer loop

  // 0) one-time (per launch) weight conversion f32 -> f16
  cvt_launch(w_in,  w_in16, (size_t)DMODEL * DMODEL, stream);
  cvt_launch(wqkv,  wqkv16, (size_t)NLAYER * QKVN * DMODEL, stream);
  cvt_launch(wo,    wo16,   (size_t)NLAYER * DMODEL * DMODEL, stream);
  cvt_launch(w1,    w116,   (size_t)NLAYER * DFFDIM * DMODEL, stream);
  cvt_launch(w2,    w216,   (size_t)NLAYER * DMODEL * DFFDIM, stream);
  cvt_launch(w_out, wout16, (size_t)SDDIM * DMODEL, stream);

  // 1) cross-speaker context + input pack + input projection
  cross_ctx_kernel<<<BATCH, 256, 0, stream>>>(delta, valid, spk, cross);
  pack_kernel<<<ROWS, 256, 0, stream>>>(delta, sctx, cross, xin16);
  gemm_kernel<<<dim3(ROWS / 64, DMODEL / 256), 256, 0, stream>>>(
      xin16, w_in16, b_in, nullptr, x0, nullptr, ROWS, DMODEL, DMODEL, 0);

  float* x = x0;
  float* xalt = x1;
  for (int l = 0; l < NLAYER; ++l) {
    ln_kernel<<<ROWS, 256, 0, stream>>>(x, ln1g + l * DMODEL, ln1b + l * DMODEL, h16, DMODEL);
    gemm_kernel<<<dim3(ROWS / 64, QKVN / 256), 256, 0, stream>>>(
        h16, wqkv16 + (size_t)l * QKVN * DMODEL, bqkv + (size_t)l * QKVN,
        nullptr, nullptr, qkv16, ROWS, QKVN, DMODEL, 0);
    vtrans_kernel<<<(ROWS * DMODEL) / 256, 256, 0, stream>>>(qkv16, Vt);
    attn_kernel<<<(BATCH * NHEAD * (TDIM / 16)) / 8, 256, 0, stream>>>(qkv16, Vt, valid, attn16);
    gemm_kernel<<<dim3(ROWS / 64, DMODEL / 256), 256, 0, stream>>>(
        attn16, wo16 + (size_t)l * DMODEL * DMODEL, bo + (size_t)l * DMODEL,
        x, xalt, nullptr, ROWS, DMODEL, DMODEL, 0);
    { float* tmp = x; x = xalt; xalt = tmp; }
    ln_kernel<<<ROWS, 256, 0, stream>>>(x, ln2g + l * DMODEL, ln2b + l * DMODEL, h16, DMODEL);
    gemm_kernel<<<dim3(ROWS / 64, DFFDIM / 256), 256, 0, stream>>>(
        h16, w116 + (size_t)l * DFFDIM * DMODEL, b1 + (size_t)l * DFFDIM,
        nullptr, nullptr, ffn16, ROWS, DFFDIM, DMODEL, 1);
    gemm_kernel<<<dim3(ROWS / 64, DMODEL / 256), 256, 0, stream>>>(
        ffn16, w216 + (size_t)l * DMODEL * DFFDIM, b2 + (size_t)l * DMODEL,
        x, xalt, nullptr, ROWS, DMODEL, DFFDIM, 0);
    { float* tmp = x; x = xalt; xalt = tmp; }
  }

  // 2) output projection + final LN + shift-right with initial state
  cvt_launch(x, x16, (size_t)ROWS * DMODEL, stream);
  gemm_kernel<<<dim3(ROWS / 64, SDDIM / 256), 256, 0, stream>>>(
      x16, wout16, b_out, nullptr, proj, nullptr, ROWS, SDDIM, DMODEL, 0);
  out_kernel<<<ROWS, 256, 0, stream>>>(proj, lnog, lnob, istate, outp);
}
